// NeuralFingerprint_89395449299452
// MI455X (gfx1250) — compile-verified, matrix operand order checked by reading, not run on hardware
//
#include <hip/hip_runtime.h>
#include <hip/hip_bf16.h>
#include <stdint.h>

// ---------------------------------------------------------------------------
// Problem constants (from reference)
// ---------------------------------------------------------------------------
#define N_ATOMS   524288
#define N_EDGES   1433600
#define N0_ATOMS  8192          // degree-0 atoms (zero neighbor contribution)
#define NODE_F    64
#define EDGE_F    16
#define HDIM      128
#define OUT_F     128
#define BN_EPS    1e-5f

// conv0 concat width 64+16=80 -> pad to 96 ; conv1 concat width 128+16=144 -> pad to 160
#define CIN0  80
#define SK0   96
#define CIN1  144
#define SK1   160

// degree segment boundaries in atom space (atoms sorted by degree); all are
// multiples of 32, so a 32-row tile never straddles a degree segment.
__device__ __forceinline__ int deg_index_of_row(int atomRow) {
    if (atomRow < 73728)  return 0;
    if (atomRow < 204800) return 1;
    if (atomRow < 401408) return 2;
    if (atomRow < 499712) return 3;
    return 4;
}

typedef __attribute__((ext_vector_type(16))) __bf16 bf16x16;
typedef __attribute__((ext_vector_type(8)))  __bf16 bf16x8;
typedef __attribute__((ext_vector_type(8)))  float  floatx8;
typedef __attribute__((ext_vector_type(4)))  unsigned int u32x4;
typedef __attribute__((ext_vector_type(8)))  int  i32x8;
typedef __attribute__((ext_vector_type(4)))  int  i32x4;

// ---------------------------------------------------------------------------
// TDM: stage a 32 x K bf16 tile from global into LDS with one tensor op.
// D# built per ISA 8.3/8.4: group0 = {count=1 | lds_addr | global_addr | type=2},
// group1 = {data_size=2B, tensor_dim0=K, tensor_dim1=32, tile_dim0=K,
//           tile_dim1=32, tensor_dim0_stride=K}.  Issued by wave 0 only
// (TDM ignores EXEC), tracked with TENSORcnt, published via barrier.
// ---------------------------------------------------------------------------
__device__ __forceinline__ void tdm_stage_A(const __bf16* __restrict__ gsrc, int K,
                                            __bf16* sA) {
    if (threadIdx.x < 32) {
        uint64_t ga  = (uint64_t)(uintptr_t)gsrc;
        uint32_t lds = (uint32_t)(uintptr_t)sA;     // low 32 bits = LDS offset
        u32x4 g0 = { 1u,                            // count=1, user descriptor
                     lds,                           // lds_addr (bits 63:32)
                     (uint32_t)ga,                  // global_addr lo
                     (uint32_t)(ga >> 32) | (2u << 30) };  // addr hi | type=2
        i32x8 g1 = {0, 0, 0, 0, 0, 0, 0, 0};
        g1[0] = (1 << 16);                          // data_size: 1 -> 2 bytes
        g1[1] = (int)((K & 0xFFFF) << 16);          // tensor_dim0[15:0] @bits63:48
        g1[2] = (int)(((unsigned)K >> 16) | (32u << 16)); // dim0 hi | tensor_dim1=32
        g1[3] = (int)((K & 0xFFFF) << 16);          // tensor_dim1 hi=0 | tile_dim0=K
        g1[4] = 32;                                 // tile_dim1=32, tile_dim2=0
        g1[5] = K;                                  // tensor_dim0_stride lo32
        i32x4 z4 = {0, 0, 0, 0};
#if defined(__clang_major__) && (__clang_major__ >= 23)
        i32x8 z8 = {0, 0, 0, 0, 0, 0, 0, 0};
        __builtin_amdgcn_tensor_load_to_lds(g0, g1, z4, z4, z8, 0);
#else
        __builtin_amdgcn_tensor_load_to_lds(g0, g1, z4, z4, 0);
#endif
        __builtin_amdgcn_s_wait_tensorcnt(0);
    }
    __syncthreads();
}

// ---------------------------------------------------------------------------
// Two stacked 16x16 output tiles (rows 0-15 / 16-31 of the LDS A tile) against
// one B fragment: per K-step, B fragment (global, 2x b128) is reused by two
// v_wmma_f32_16x16x32_bf16; A fragments come from LDS (ds_load_b128).
// Fragment layout per ISA 7.12.2 (two 16B chunks split at the half-wave).
// ---------------------------------------------------------------------------
__device__ __forceinline__ void
wmma_tiles2_bf16(const __bf16* sA, int K,
                 const __bf16* __restrict__ Btile, int ldb, int lane,
                 floatx8& c0, floatx8& c1)
{
    const int half = (lane >> 4) & 1;
    const int idx  = lane & 15;
    const __bf16* a0 = sA + (size_t)idx * K + 8 * half;
    const __bf16* a1 = sA + (size_t)(16 + idx) * K + 8 * half;
    const __bf16* br = Btile + (size_t)idx * ldb + 8 * half;
    for (int ks = 0; ks < K; ks += 32) {
        union { bf16x16 v; bf16x8 h[2]; } au0, au1, bu;
        bu.h[0]  = *(const bf16x8*)(br + ks);
        bu.h[1]  = *(const bf16x8*)(br + ks + 16);
        au0.h[0] = *(const bf16x8*)(a0 + ks);
        au0.h[1] = *(const bf16x8*)(a0 + ks + 16);
        au1.h[0] = *(const bf16x8*)(a1 + ks);
        au1.h[1] = *(const bf16x8*)(a1 + ks + 16);
        c0 = __builtin_amdgcn_wmma_f32_16x16x32_bf16(
                 false, au0.v, false, bu.v, (short)0, c0, false, false);
        c1 = __builtin_amdgcn_wmma_f32_16x16x32_bf16(
                 false, au1.v, false, bu.v, (short)0, c1, false, false);
    }
}

// ---------------------------------------------------------------------------
// elementwise f32 -> bf16
// ---------------------------------------------------------------------------
__global__ __launch_bounds__(256) void k_cvt_bf16(const float* __restrict__ src,
                                                  __bf16* __restrict__ dst, int n) {
    int i = blockIdx.x * 256 + threadIdx.x;
    if (i < n) dst[i] = (__bf16)src[i];
}

// pack degW [5][HDIM][CIN] f32 -> [5][HDIM][SK] bf16, zero padded
__global__ __launch_bounds__(256) void k_pack_degW(const float* __restrict__ src,
                                                   __bf16* __restrict__ dst,
                                                   int CIN, int SK) {
    int i = blockIdx.x * 256 + threadIdx.x;
    int total = 5 * HDIM * SK;
    if (i >= total) return;
    int c   = i % SK;
    int row = i / SK;
    dst[i] = (c < CIN) ? (__bf16)src[(size_t)row * CIN + c] : (__bf16)0.f;
}

// ---------------------------------------------------------------------------
// per-degree gather+sum -> packed bf16 neighbor matrix gb[(atom-N0)][SK]
// ---------------------------------------------------------------------------
__global__ __launch_bounds__(256) void k_gather(const __bf16* __restrict__ nodeB, int W,
                                                const __bf16* __restrict__ edgeB,
                                                const int* __restrict__ nbrN,
                                                const int* __restrict__ nbrE,
                                                __bf16* __restrict__ gb,
                                                int n_d, int d, int SK, int rowoff) {
    int i = blockIdx.x * 256 + threadIdx.x;
    int total = n_d * SK;
    if (i >= total) return;
    int c   = i % SK;
    int row = i / SK;
    float s = 0.f;
    if (c < W) {
        for (int j = 0; j < d; ++j) {
            int nb = nbrN[row * d + j];
            s += (float)nodeB[(size_t)nb * W + c];
        }
    } else if (c < W + EDGE_F) {
        int ec = c - W;
        for (int j = 0; j < d; ++j) {
            int eb = nbrE[row * d + j];
            s += (float)edgeB[(size_t)eb * EDGE_F + ec];
        }
    }
    gb[(size_t)(rowoff + row) * SK + c] = (__bf16)s;
}

// ---------------------------------------------------------------------------
// act[m, :] = X @ selfW^T + bias      block: 32 rows x 128 cols, 8 waves,
// A tile staged by TDM into LDS, each wave 2 stacked 16x16 tiles.
// ---------------------------------------------------------------------------
__global__ __launch_bounds__(256) void k_gemm_self(const __bf16* __restrict__ X, int K,
                                                   const __bf16* __restrict__ W,
                                                   const float* __restrict__ bias,
                                                   float* __restrict__ act) {
    __shared__ __bf16 sA[32 * HDIM];          // K <= 128
    int m0 = blockIdx.x * 32;
    tdm_stage_A(X + (size_t)m0 * K, K, sA);

    int wave = threadIdx.x >> 5;
    int lane = threadIdx.x & 31;
    int n0   = wave * 16;
    floatx8 c0 = {0.f,0.f,0.f,0.f,0.f,0.f,0.f,0.f};
    floatx8 c1 = c0;
    wmma_tiles2_bf16(sA, K, W + (size_t)n0 * K, K, lane, c0, c1);

    int half = lane >> 4, nn = lane & 15;
    float bv = bias[n0 + nn];
    #pragma unroll
    for (int r = 0; r < 8; ++r) {
        act[(size_t)(m0 + 8 * half + r) * HDIM + n0 + nn]      = c0[r] + bv;
        act[(size_t)(m0 + 16 + 8 * half + r) * HDIM + n0 + nn] = c1[r] + bv;
    }
}

// act[N0+row, :] += gb @ degW[deg]^T   (rows >= N0; degree from row index)
__global__ __launch_bounds__(256) void k_gemm_nbr(const __bf16* __restrict__ GB, int SK,
                                                  const __bf16* __restrict__ degWp,
                                                  float* __restrict__ act) {
    __shared__ __bf16 sA[32 * SK1];           // SK <= 160
    int gbrow0 = blockIdx.x * 32;
    int atom0  = N0_ATOMS + gbrow0;
    int di     = deg_index_of_row(atom0);     // whole 32-row tile has one degree
    tdm_stage_A(GB + (size_t)gbrow0 * SK, SK, sA);

    int wave = threadIdx.x >> 5;
    int lane = threadIdx.x & 31;
    int n0   = wave * 16;
    const __bf16* Wt = degWp + ((size_t)di * HDIM + n0) * SK;
    floatx8 c0 = {0.f,0.f,0.f,0.f,0.f,0.f,0.f,0.f};
    floatx8 c1 = c0;
    wmma_tiles2_bf16(sA, SK, Wt, SK, lane, c0, c1);

    int half = lane >> 4, nn = lane & 15;
    #pragma unroll
    for (int r = 0; r < 8; ++r) {
        size_t o0 = (size_t)(atom0 + 8 * half + r) * HDIM + n0 + nn;
        size_t o1 = (size_t)(atom0 + 16 + 8 * half + r) * HDIM + n0 + nn;
        act[o0] += c0[r];
        act[o1] += c1[r];
    }
}

// ---------------------------------------------------------------------------
// BatchNorm: deterministic two-pass. 256 blocks of partials, then one reduce.
// ---------------------------------------------------------------------------
__global__ __launch_bounds__(256) void k_bn_part(const float* __restrict__ act,
                                                 float* __restrict__ part) {
    const int ROWS = N_ATOMS / 256;
    int b = blockIdx.x, t = threadIdx.x;
    int col = t & 127, rs = t >> 7;
    size_t base = (size_t)b * ROWS;
    float s = 0.f, s2 = 0.f;
    for (int i = rs; i < ROWS; i += 2) {
        float v = act[(base + i) * HDIM + col];
        s += v; s2 += v * v;
    }
    __shared__ float ls[256], ls2[256];
    ls[t] = s; ls2[t] = s2;
    __syncthreads();
    if (t < 128) {
        part[(size_t)b * 256 + t]       = ls[t] + ls[t + 128];
        part[(size_t)b * 256 + 128 + t] = ls2[t] + ls2[t + 128];
    }
}

__global__ __launch_bounds__(128) void k_bn_final(const float* __restrict__ part,
                                                  float* __restrict__ stats) {
    int t = threadIdx.x;
    float S = 0.f, S2 = 0.f;
    for (int b = 0; b < 256; ++b) {
        S  += part[(size_t)b * 256 + t];
        S2 += part[(size_t)b * 256 + 128 + t];
    }
    float mean = S / (float)N_ATOMS;
    float var  = S2 / (float)N_ATOMS - mean * mean;
    stats[t]       = mean;
    stats[128 + t] = rsqrtf(var + BN_EPS);
}

__global__ __launch_bounds__(256) void k_bn_relu(const float* __restrict__ act,
                                                 const float* __restrict__ stats,
                                                 __bf16* __restrict__ xh) {
    int i = blockIdx.x * 256 + threadIdx.x;
    int col = i & 127;
    float v = (act[i] - stats[col]) * stats[128 + col];
    xh[i] = (__bf16)fmaxf(v, 0.f);
}

// ---------------------------------------------------------------------------
// acc (+)= softmax(X @ W^T + b)   block: 32 rows x 128 cols; TDM-staged A;
// WMMA -> LDS logits -> shfl row-softmax (8-lane groups) -> global write/acc.
// ---------------------------------------------------------------------------
__global__ __launch_bounds__(256) void k_softmax_acc(const __bf16* __restrict__ X, int K,
                                                     const __bf16* __restrict__ W,
                                                     const float* __restrict__ b,
                                                     float* __restrict__ out, int add) {
    __shared__ __bf16 sA[32 * HDIM];          // K <= 128
    __shared__ float  sl[32][128];
    int m0 = blockIdx.x * 32;
    tdm_stage_A(X + (size_t)m0 * K, K, sA);

    int wave = threadIdx.x >> 5;
    int lane = threadIdx.x & 31;
    int n0   = wave * 16;
    floatx8 c0 = {0.f,0.f,0.f,0.f,0.f,0.f,0.f,0.f};
    floatx8 c1 = c0;
    wmma_tiles2_bf16(sA, K, W + (size_t)n0 * K, K, lane, c0, c1);

    int half = lane >> 4, nn = lane & 15;
    float bv = b[n0 + nn];
    #pragma unroll
    for (int r = 0; r < 8; ++r) {
        sl[8 * half + r][n0 + nn]      = c0[r] + bv;
        sl[16 + 8 * half + r][n0 + nn] = c1[r] + bv;
    }
    __syncthreads();

    int t = threadIdx.x;
    int r = t >> 3;          // row 0..31
    int g = t & 7;           // 8 threads per row, 16 cols each
    float v[16];
    #pragma unroll
    for (int j = 0; j < 16; ++j) v[j] = sl[r][g * 16 + j];
    float m = v[0];
    #pragma unroll
    for (int j = 1; j < 16; ++j) m = fmaxf(m, v[j]);
    for (int mask = 1; mask < 8; mask <<= 1)
        m = fmaxf(m, __shfl_xor(m, mask, 8));
    float s = 0.f;
    #pragma unroll
    for (int j = 0; j < 16; ++j) { v[j] = __expf(v[j] - m); s += v[j]; }
    for (int mask = 1; mask < 8; mask <<= 1)
        s += __shfl_xor(s, mask, 8);
    float inv = 1.0f / s;
    float* orow = out + (size_t)(m0 + r) * OUT_F + g * 16;
    if (add) {
        #pragma unroll
        for (int j = 0; j < 16; ++j) orow[j] += v[j] * inv;
    } else {
        #pragma unroll
        for (int j = 0; j < 16; ++j) orow[j] = v[j] * inv;
    }
}

// ---------------------------------------------------------------------------
// host launcher
// ---------------------------------------------------------------------------
extern "C" void kernel_launch(void* const* d_in, const int* in_sizes, int n_in,
                              void* d_out, int out_size, void* d_ws, size_t ws_size,
                              hipStream_t stream) {
    (void)in_sizes; (void)n_in; (void)out_size; (void)ws_size;

    const float* node_feat = (const float*)d_in[0];
    const float* edge_feat = (const float*)d_in[1];
    const int* nbrN[5] = { (const int*)d_in[2], (const int*)d_in[4], (const int*)d_in[6],
                           (const int*)d_in[8], (const int*)d_in[10] };
    const int* nbrE[5] = { (const int*)d_in[3], (const int*)d_in[5], (const int*)d_in[7],
                           (const int*)d_in[9], (const int*)d_in[11] };
    const float* out_W0 = (const float*)d_in[13]; const float* out_b0 = (const float*)d_in[14];
    const float* out_W1 = (const float*)d_in[15]; const float* out_b1 = (const float*)d_in[16];
    const float* out_W2 = (const float*)d_in[17]; const float* out_b2 = (const float*)d_in[18];
    const float* c0_selfW = (const float*)d_in[19];
    const float* c0_degW  = (const float*)d_in[20];
    const float* c0_bias  = (const float*)d_in[21];
    const float* c1_selfW = (const float*)d_in[22];
    const float* c1_degW  = (const float*)d_in[23];
    const float* c1_bias  = (const float*)d_in[24];
    float* out = (float*)d_out;

    // workspace carve-up (256B aligned)
    char* wp = (char*)d_ws;
    auto carve = [&](size_t bytes) -> char* {
        char* p = wp; wp += (bytes + 255) & ~(size_t)255; return p;
    };
    __bf16* xb0  = (__bf16*)carve((size_t)N_ATOMS * NODE_F * 2);
    __bf16* eb   = (__bf16*)carve((size_t)N_EDGES * EDGE_F * 2);
    __bf16* xh   = (__bf16*)carve((size_t)N_ATOMS * HDIM * 2);
    float*  act  = (float*) carve((size_t)N_ATOMS * HDIM * 4);
    __bf16* gb   = (__bf16*)carve((size_t)(N_ATOMS - N0_ATOMS) * SK1 * 2);
    float*  part = (float*) carve(256 * 256 * 4);
    float*  stats= (float*) carve(256 * 4);
    __bf16* w0b  = (__bf16*)carve(HDIM * NODE_F * 2);
    __bf16* w1b  = (__bf16*)carve(HDIM * HDIM * 2);
    __bf16* w2b  = (__bf16*)carve(HDIM * HDIM * 2);
    __bf16* s0b  = (__bf16*)carve(HDIM * NODE_F * 2);
    __bf16* s1b  = (__bf16*)carve(HDIM * HDIM * 2);
    __bf16* d0p  = (__bf16*)carve(5 * HDIM * SK0 * 2);
    __bf16* d1p  = (__bf16*)carve(5 * HDIM * SK1 * 2);

    auto blocks = [](int n) { return (n + 255) / 256; };

    // --- convert data + weights to bf16 ---
    k_cvt_bf16<<<blocks(N_ATOMS * NODE_F), 256, 0, stream>>>(node_feat, xb0, N_ATOMS * NODE_F);
    k_cvt_bf16<<<blocks(N_EDGES * EDGE_F), 256, 0, stream>>>(edge_feat, eb, N_EDGES * EDGE_F);
    k_cvt_bf16<<<blocks(HDIM * NODE_F), 256, 0, stream>>>(out_W0, w0b, HDIM * NODE_F);
    k_cvt_bf16<<<blocks(HDIM * HDIM), 256, 0, stream>>>(out_W1, w1b, HDIM * HDIM);
    k_cvt_bf16<<<blocks(HDIM * HDIM), 256, 0, stream>>>(out_W2, w2b, HDIM * HDIM);
    k_cvt_bf16<<<blocks(HDIM * NODE_F), 256, 0, stream>>>(c0_selfW, s0b, HDIM * NODE_F);
    k_cvt_bf16<<<blocks(HDIM * HDIM), 256, 0, stream>>>(c1_selfW, s1b, HDIM * HDIM);
    k_pack_degW<<<blocks(5 * HDIM * SK0), 256, 0, stream>>>(c0_degW, d0p, CIN0, SK0);
    k_pack_degW<<<blocks(5 * HDIM * SK1), 256, 0, stream>>>(c1_degW, d1p, CIN1, SK1);

    const int degCnt[5] = { 65536, 131072, 196608, 98304, 24576 };
    const int degOff[5] = { 0, 65536, 196608, 393216, 491520 };

    // --- head 0: acc = softmax(node_feat @ W0^T + b0) ---
    k_softmax_acc<<<N_ATOMS / 32, 256, 0, stream>>>(xb0, NODE_F, w0b, out_b0, out, 0);

    // --- conv0 ---
    for (int d = 1; d <= 5; ++d) {
        int n_d = degCnt[d - 1];
        k_gather<<<blocks(n_d * SK0), 256, 0, stream>>>(xb0, NODE_F, eb,
            nbrN[d - 1], nbrE[d - 1], gb, n_d, d, SK0, degOff[d - 1]);
    }
    k_gemm_self<<<N_ATOMS / 32, 256, 0, stream>>>(xb0, NODE_F, s0b, c0_bias, act);
    k_gemm_nbr<<<(N_ATOMS - N0_ATOMS) / 32, 256, 0, stream>>>(gb, SK0, d0p, act);
    k_bn_part<<<256, 256, 0, stream>>>(act, part);
    k_bn_final<<<1, 128, 0, stream>>>(part, stats);
    k_bn_relu<<<blocks(N_ATOMS * HDIM), 256, 0, stream>>>(act, stats, xh);

    // --- head 1: acc += softmax(x1 @ W1^T + b1) ---
    k_softmax_acc<<<N_ATOMS / 32, 256, 0, stream>>>(xh, HDIM, w1b, out_b1, out, 1);

    // --- conv1 (gathers read xh before it is overwritten by the next bn_relu) ---
    for (int d = 1; d <= 5; ++d) {
        int n_d = degCnt[d - 1];
        k_gather<<<blocks(n_d * SK1), 256, 0, stream>>>(xh, HDIM, eb,
            nbrN[d - 1], nbrE[d - 1], gb, n_d, d, SK1, degOff[d - 1]);
    }
    k_gemm_self<<<N_ATOMS / 32, 256, 0, stream>>>(xh, HDIM, s1b, c1_bias, act);
    k_gemm_nbr<<<(N_ATOMS - N0_ATOMS) / 32, 256, 0, stream>>>(gb, SK1, d1p, act);
    k_bn_part<<<256, 256, 0, stream>>>(act, part);
    k_bn_final<<<1, 128, 0, stream>>>(part, stats);
    k_bn_relu<<<blocks(N_ATOMS * HDIM), 256, 0, stream>>>(act, stats, xh);

    // --- head 2: acc += softmax(x2 @ W2^T + b2)  (atom_index is arange -> identity) ---
    k_softmax_acc<<<N_ATOMS / 32, 256, 0, stream>>>(xh, HDIM, w2b, out_b2, out, 1);
}
